// SimpleTransformerLayer_37958920962066
// MI455X (gfx1250) — compile-verified, matrix operand order checked by reading, not run on hardware
//
#include <hip/hip_runtime.h>
#include <hip/hip_bf16.h>
#include <math.h>

// ---------------- problem constants ----------------
#define D_MODEL 1024
#define NUM_HEADS 16
#define D_K 64
#define D_FF 4096
#define SEQ 2048
#define BATCH 2
#define MTOK (BATCH * SEQ)   // 4096 rows

typedef __attribute__((ext_vector_type(16))) __bf16 v16bf;
typedef __attribute__((ext_vector_type(8)))  float  v8f;
typedef __attribute__((ext_vector_type(4)))  int    v4i;
typedef __attribute__((ext_vector_type(4)))  unsigned int u32x4;
typedef unsigned short u16;
typedef unsigned int   u32;

typedef __attribute__((address_space(1))) v4i as1_v4i;   // global int4*
typedef __attribute__((address_space(3))) v4i as3_v4i;   // LDS int4*

// ---- CDNA5 async global->LDS path (ASYNCcnt), with safe fallback ----
#if defined(__has_builtin)
#  if __has_builtin(__builtin_amdgcn_global_load_async_to_lds_b128)
#    define HAVE_ASYNC 1
#  else
#    define HAVE_ASYNC 0
#  endif
#else
#  define HAVE_ASYNC 0
#endif

#if HAVE_ASYNC
#  define ASYNC_COPY16(gsrc, ldst) \
     __builtin_amdgcn_global_load_async_to_lds_b128((as1_v4i*)(gsrc), (as3_v4i*)(ldst), 0, 0)
#  if __has_builtin(__builtin_amdgcn_s_wait_asynccnt)
#    define WAIT_ASYNC(n) __builtin_amdgcn_s_wait_asynccnt(n)
#  else
#    define WAIT_ASYNC(n) asm volatile("s_wait_asynccnt %0" ::"i"(n) : "memory")
#  endif
#else
#  define ASYNC_COPY16(gsrc, ldst) (*(u32x4*)(ldst) = *(const u32x4*)(gsrc))
#  define WAIT_ASYNC(n) do {} while (0)
#endif

// ---------------- bf16 helpers ----------------
__device__ __forceinline__ u16 f2bf(float f) {
    u32 u = __float_as_uint(f);
    u32 r = u + 0x7FFFu + ((u >> 16) & 1u);   // round-to-nearest-even
    return (u16)(r >> 16);
}
__device__ __forceinline__ float bf2f(u16 s) {
    return __uint_as_float(((u32)s) << 16);
}

// ---------------- WMMA fragment loader ----------------
// 16-bit A fragment (16x32, MxK) per ISA 7.12.2:
//   lane L: row M = L%16 ; half = L/16
//   VGPR v: K = (v/4)*16 + half*8 + (v%4)*2 (+k0), two bf16 per dword.
// B fragment (32x16 KxN) is the transpose mirror (lane holds column N = L%16,
// same K striping), so rows of W (for y = x @ W^T) load with the same loader.
__device__ __forceinline__ v16bf load_frag(const u16* rowptr, int half, int k0) {
    union { u32 u[8]; v16bf v; } r;
#pragma unroll
    for (int q = 0; q < 8; ++q) {
        int kb = k0 + ((q >> 2) << 4) + (half << 3) + ((q & 3) << 1);
        r.u[q] = *(const u32*)(rowptr + kb);
    }
    return r.v;
}

__device__ __forceinline__ v8f wmma_bf16(v16bf a, v16bf b, v8f c) {
    return __builtin_amdgcn_wmma_f32_16x16x32_bf16(
        false, a, false, b, (short)0, c, false, false);
}

// ---------------- f32 -> bf16 convert ----------------
__global__ void k_f32_to_bf16(const float* __restrict__ in, u16* __restrict__ out, int n) {
    for (int i = blockIdx.x * blockDim.x + threadIdx.x; i < n; i += gridDim.x * blockDim.x)
        out[i] = f2bf(in[i]);
}

// ---------------- LDS-staged WMMA GEMM: Y = A(bf16) @ W(bf16)^T --------------
// Block = 256 threads = 8 waves, tile = 128(M) x 64(N), K staged 64 at a time,
// double-buffered async global->LDS.  Rows padded to 72 u16 (144B = 9*16B).
#define LDP 72
#define MODE_F32       0   // float C[m*Ncols + n]
#define MODE_QK_BF16   1   // bf16 C[((b*H + h)*SEQ + t)*DK + d]
#define MODE_V_BF16    2   // bf16 C[((b*H + h)*DK + d)*SEQ + t]
#define MODE_GELU_BF16 3   // bf16 gelu(C)[m*Ncols + n]

__device__ __forceinline__ void stage_copy(const u16* Ag, int lda, const u16* Wg, int ldw,
                                           u16* sa, u16* sb, int k0, int tid) {
#pragma unroll
    for (int i = 0; i < 4; ++i) {              // A panel: 128 rows x 64 k = 1024 x 16B
        int c = tid + i * 256;
        int row = c >> 3, cc = (c & 7) << 3;
        ASYNC_COPY16(Ag + (size_t)row * lda + k0 + cc, sa + row * LDP + cc);
    }
#pragma unroll
    for (int i = 0; i < 2; ++i) {              // B panel: 64 rows x 64 k = 512 x 16B
        int c = tid + i * 256;
        int row = c >> 3, cc = (c & 7) << 3;
        ASYNC_COPY16(Wg + (size_t)row * ldw + k0 + cc, sb + row * LDP + cc);
    }
}

__global__ void k_gemm_wmma(const u16* __restrict__ A, int lda,
                            const u16* __restrict__ W, int ldw,
                            void* __restrict__ Cv,
                            int tilesN, int K, int Ncols, int mode) {
    alignas(16) __shared__ u16 s_a[2][128 * LDP];   // 36 KB
    alignas(16) __shared__ u16 s_b[2][64 * LDP];    // 18 KB

    const int tid  = threadIdx.x;
    const int wave = tid >> 5;
    const int lane = tid & 31;
    const int half = lane >> 4;
    const int cn   = lane & 15;
    const int mTile = blockIdx.x / tilesN;
    const int nTile = blockIdx.x % tilesN;

    const u16* Ag = A + (size_t)mTile * 128 * lda;
    const u16* Wg = W + (size_t)nTile * 64 * ldw;

    v8f acc[4];
#pragma unroll
    for (int t = 0; t < 4; ++t) acc[t] = (v8f){0.f,0.f,0.f,0.f,0.f,0.f,0.f,0.f};

    const int stages = K >> 6;
    int buf = 0;
    stage_copy(Ag, lda, Wg, ldw, &s_a[0][0], &s_b[0][0], 0, tid);

    for (int s = 0; s < stages; ++s) {
        if (s + 1 < stages) {
            stage_copy(Ag, lda, Wg, ldw, &s_a[buf ^ 1][0], &s_b[buf ^ 1][0], (s + 1) << 6, tid);
            WAIT_ASYNC(6);    // stage-s loads done; next stage's 6 remain in flight
        } else {
            WAIT_ASYNC(0);
        }
        __syncthreads();      // every wave's stage-s panel landed

        const u16* sa  = &s_a[buf][0] + (wave * 16 + cn) * LDP;
        const u16* sb0 = &s_b[buf][0] + cn * LDP;

        // prefetch the whole stage's fragments into distinct registers so the
        // scheduler can hoist all ds_load_b128 ahead of the WMMA chain
        v16bf a0 = load_frag(sa, half, 0);
        v16bf a1 = load_frag(sa, half, 32);
        v16bf b0[4], b1[4];
#pragma unroll
        for (int t = 0; t < 4; ++t) b0[t] = load_frag(sb0 + t * 16 * LDP, half, 0);
#pragma unroll
        for (int t = 0; t < 4; ++t) b1[t] = load_frag(sb0 + t * 16 * LDP, half, 32);
#pragma unroll
        for (int t = 0; t < 4; ++t) acc[t] = wmma_bf16(a0, b0[t], acc[t]);
#pragma unroll
        for (int t = 0; t < 4; ++t) acc[t] = wmma_bf16(a1, b1[t], acc[t]);

        __syncthreads();      // panel fully consumed before it is overwritten
        buf ^= 1;
    }

    // epilogue: lane holds N = cn, M = r + 8*half (wave supplies 16-row slice)
#pragma unroll
    for (int t = 0; t < 4; ++t) {
#pragma unroll
        for (int r = 0; r < 8; ++r) {
            int m = mTile * 128 + wave * 16 + r + 8 * half;
            int n = nTile * 64 + t * 16 + cn;
            float v = acc[t][r];
            if (mode == MODE_F32) {
                ((float*)Cv)[(size_t)m * Ncols + n] = v;
            } else if (mode == MODE_QK_BF16) {
                size_t idx = (((size_t)((m >> 11) * NUM_HEADS + (n >> 6)) * SEQ) + (m & (SEQ - 1))) * D_K + (n & 63);
                ((u16*)Cv)[idx] = f2bf(v);
            } else if (mode == MODE_V_BF16) {
                size_t idx = ((size_t)((m >> 11) * NUM_HEADS + (n >> 6)) * D_K + (n & 63)) * SEQ + (m & (SEQ - 1));
                ((u16*)Cv)[idx] = f2bf(v);
            } else { // MODE_GELU_BF16 (exact gelu)
                float g = 0.5f * v * (1.0f + erff(v * 0.70710678118654752f));
                ((u16*)Cv)[(size_t)m * Ncols + n] = f2bf(g);
            }
        }
    }
}

// ---------------- attention: one WG per (b*H+h, 16-row block) ----------------
// dyn LDS: u16 s_attn[16*SEQ] (64KB) | float s_out[16*64] | float s_red[256] | float s_sum[16]
__global__ void k_attention(const u16* __restrict__ q, const u16* __restrict__ k,
                            const u16* __restrict__ vt, u16* __restrict__ attnout) {
    extern __shared__ char smem[];
    u16*   s_attn = (u16*)smem;                         // 16 x SEQ bf16 scores/weights
    float* s_out  = (float*)(smem + 16 * SEQ * 2);      // 16 x 64 accum
    float* s_red  = (float*)(smem + 16 * SEQ * 2 + 16 * 64 * 4);
    float* s_sum  = s_red + 256;

    const int tid  = threadIdx.x;
    const int wave = tid >> 5;
    const int lane = tid & 31;
    const int half = lane >> 4;
    const int cn   = lane & 15;
    const int mblk = blockIdx.x;           // 0..127
    const int bh   = blockIdx.y;           // 0..31  (= b*H + h)

    for (int i = tid; i < 16 * 64; i += blockDim.x) s_out[i] = 0.f;

    const u16* qbase = q  + (size_t)bh * SEQ * D_K;     // [t][d]
    const u16* kbase = k  + (size_t)bh * SEQ * D_K;     // [t][d]
    const u16* vbase = vt + (size_t)bh * D_K * SEQ;     // [d][t]

    // ---- phase 1: scores[16][SEQ] = (Q_tile @ K^T) * 1/sqrt(Dk), bf16 in LDS
    const u16* arow = qbase + (size_t)(mblk * 16 + cn) * D_K;
    for (int j = 0; j < 16; ++j) {
        int n0 = (wave + j * 8) * 16;                   // 8 waves x 16 tiles = 128 tiles
        v8f c = (v8f){0.f,0.f,0.f,0.f,0.f,0.f,0.f,0.f};
#pragma unroll
        for (int k0 = 0; k0 < D_K; k0 += 32) {
            v16bf a = load_frag(arow, half, k0);
            v16bf b = load_frag(kbase + (size_t)(n0 + cn) * D_K, half, k0);
            c = wmma_bf16(a, b, c);
        }
#pragma unroll
        for (int r = 0; r < 8; ++r) {
            int m = r + 8 * half;
            s_attn[m * SEQ + n0 + cn] = f2bf(c[r] * 0.125f);   // 1/sqrt(64)
        }
    }
    __syncthreads();

    // ---- phase 2: row softmax (16 rows, 16 threads per row)
    const int row = tid >> 4;
    const int c16 = tid & 15;
    float mx = -3.0e38f;
    for (int j = 0; j < SEQ / 16; ++j)
        mx = fmaxf(mx, bf2f(s_attn[row * SEQ + c16 + (j << 4)]));
    s_red[tid] = mx;
    __syncthreads();
    for (int st = 8; st > 0; st >>= 1) {
        if (c16 < st) s_red[tid] = fmaxf(s_red[tid], s_red[tid + st]);
        __syncthreads();
    }
    mx = s_red[row << 4];
    __syncthreads();
    float sum = 0.f;
    for (int j = 0; j < SEQ / 16; ++j) {
        int idx = row * SEQ + c16 + (j << 4);
        float p = __expf(bf2f(s_attn[idx]) - mx);
        s_attn[idx] = f2bf(p);
        sum += p;
    }
    s_red[tid] = sum;
    __syncthreads();
    for (int st = 8; st > 0; st >>= 1) {
        if (c16 < st) s_red[tid] += s_red[tid + st];
        __syncthreads();
    }
    if (c16 == 0) s_sum[row] = s_red[tid];
    __syncthreads();

    // ---- phase 3: out[16][64] = attn @ V ; each wave takes 1/8 of K=SEQ
    v8f acc[4];
#pragma unroll
    for (int t = 0; t < 4; ++t) acc[t] = (v8f){0.f,0.f,0.f,0.f,0.f,0.f,0.f,0.f};
    for (int i = 0; i < 8; ++i) {
        int k0 = wave * (SEQ / 8) + i * 32;
        v16bf a = load_frag(s_attn + (size_t)cn * SEQ, half, k0);
#pragma unroll
        for (int t = 0; t < 4; ++t) {
            v16bf b = load_frag(vbase + (size_t)(t * 16 + cn) * SEQ, half, k0);
            acc[t] = wmma_bf16(a, b, acc[t]);
        }
    }
#pragma unroll
    for (int t = 0; t < 4; ++t)
#pragma unroll
        for (int r = 0; r < 8; ++r)
            atomicAdd(&s_out[(r + 8 * half) * 64 + t * 16 + cn], acc[t][r]);   // ds_add_f32
    __syncthreads();

    // ---- write normalized tile to attnout[(b*SEQ+t)*D_MODEL + h*64 + d]
    const int b = bh >> 4, h = bh & 15;
    for (int i = tid; i < 16 * 64; i += blockDim.x) {
        int m = i >> 6, d = i & 63;
        float val = s_out[i] / s_sum[m];
        size_t t = (size_t)mblk * 16 + m;
        attnout[((size_t)b * SEQ + t) * D_MODEL + h * 64 + d] = f2bf(val);
    }
}

// ---------------- residual + LayerNorm (one block per row of D_MODEL) ----------
__global__ void k_resid_ln(const float* __restrict__ x, const float* __restrict__ d,
                           const float* __restrict__ g, const float* __restrict__ bta,
                           float* __restrict__ outf, u16* __restrict__ outbf) {
    __shared__ float s[256];
    const int tid = threadIdx.x;
    const size_t base = (size_t)blockIdx.x * D_MODEL;
    float vals[D_MODEL / 256];
    float sum = 0.f;
#pragma unroll
    for (int j = 0; j < D_MODEL / 256; ++j) {
        int i = tid + j * 256;
        vals[j] = x[base + i] + d[base + i];
        sum += vals[j];
    }
    s[tid] = sum; __syncthreads();
    for (int st = 128; st > 0; st >>= 1) { if (tid < st) s[tid] += s[tid + st]; __syncthreads(); }
    const float mean = s[0] * (1.0f / D_MODEL);
    __syncthreads();
    float vsum = 0.f;
#pragma unroll
    for (int j = 0; j < D_MODEL / 256; ++j) { float c = vals[j] - mean; vsum += c * c; }
    s[tid] = vsum; __syncthreads();
    for (int st = 128; st > 0; st >>= 1) { if (tid < st) s[tid] += s[tid + st]; __syncthreads(); }
    const float rstd = rsqrtf(s[0] * (1.0f / D_MODEL) + 1e-5f);
#pragma unroll
    for (int j = 0; j < D_MODEL / 256; ++j) {
        int i = tid + j * 256;
        float y = (vals[j] - mean) * rstd * g[i] + bta[i];
        outf[base + i] = y;
        if (outbf) outbf[base + i] = f2bf(y);
    }
}

// ---------------- host-side orchestration ----------------
extern "C" void kernel_launch(void* const* d_in, const int* in_sizes, int n_in,
                              void* d_out, int out_size, void* d_ws, size_t ws_size,
                              hipStream_t stream) {
    (void)in_sizes; (void)n_in; (void)out_size; (void)ws_size;
    const float* x  = (const float*)d_in[0];
    const float* Wq = (const float*)d_in[1];
    const float* Wk = (const float*)d_in[2];
    const float* Wv = (const float*)d_in[3];
    const float* Wo = (const float*)d_in[4];
    const float* W1 = (const float*)d_in[5];
    const float* W2 = (const float*)d_in[6];
    const float* g1 = (const float*)d_in[7];
    const float* b1 = (const float*)d_in[8];
    const float* g2 = (const float*)d_in[9];
    const float* b2 = (const float*)d_in[10];

    char* ws = (char*)d_ws;
    u16* x_bf  = (u16*)(ws + 0);                    //  8 MB  [4096][1024]
    u16* Wq_bf = (u16*)(ws + 8388608);              //  2 MB
    u16* Wk_bf = (u16*)(ws + 10485760);
    u16* Wv_bf = (u16*)(ws + 12582912);
    u16* Wo_bf = (u16*)(ws + 14680064);
    u16* W1_bf = (u16*)(ws + 16777216);             //  8 MB [4096][1024]
    u16* W2_bf = (u16*)(ws + 25165824);             //  8 MB [1024][4096]
    u16* q_bf  = (u16*)(ws + 33554432);             //  8 MB [b][h][t][d]
    u16* k_bf  = (u16*)(ws + 41943040);             //  8 MB [b][h][t][d]
    u16* v_bf  = (u16*)(ws + 50331648);             //  8 MB [b][h][d][t]
    u16* a_bf  = (u16*)(ws + 58720256);             //  8 MB attn out [tok][1024]
    u16* ff_bf = (u16*)(ws + 33554432);             // 32 MB, ALIASES q/k/v/a (dead by then)
    float* out1 = (float*)(ws + 67108864);          // 16 MB fp32 [4096][1024]
    float* out2 = out1;                             // alias (out1 dead after LN1)
    float* x2f  = (float*)(ws + 83886080);          // 16 MB fp32
    u16* x2_bf  = (u16*)(ws + 100663296);           //  8 MB

    // 1) convert inputs/weights to bf16
    k_f32_to_bf16<<<1024, 256, 0, stream>>>(x,  x_bf,  MTOK * D_MODEL);
    k_f32_to_bf16<<<512,  256, 0, stream>>>(Wq, Wq_bf, D_MODEL * D_MODEL);
    k_f32_to_bf16<<<512,  256, 0, stream>>>(Wk, Wk_bf, D_MODEL * D_MODEL);
    k_f32_to_bf16<<<512,  256, 0, stream>>>(Wv, Wv_bf, D_MODEL * D_MODEL);
    k_f32_to_bf16<<<512,  256, 0, stream>>>(Wo, Wo_bf, D_MODEL * D_MODEL);
    k_f32_to_bf16<<<1024, 256, 0, stream>>>(W1, W1_bf, D_FF * D_MODEL);
    k_f32_to_bf16<<<1024, 256, 0, stream>>>(W2, W2_bf, D_MODEL * D_FF);

    // 2) Q/K/V projections: tiles = (4096/128)*(1024/64) = 32*16 = 512 blocks
    k_gemm_wmma<<<512, 256, 0, stream>>>(x_bf, D_MODEL, Wq_bf, D_MODEL, q_bf,
                                         16, D_MODEL, D_MODEL, MODE_QK_BF16);
    k_gemm_wmma<<<512, 256, 0, stream>>>(x_bf, D_MODEL, Wk_bf, D_MODEL, k_bf,
                                         16, D_MODEL, D_MODEL, MODE_QK_BF16);
    k_gemm_wmma<<<512, 256, 0, stream>>>(x_bf, D_MODEL, Wv_bf, D_MODEL, v_bf,
                                         16, D_MODEL, D_MODEL, MODE_V_BF16);

    // 3) attention (softmax(QK^T/sqrt(dk)) @ V), ~69.5 KB dynamic LDS
    const size_t smem = (size_t)16 * SEQ * 2 + 16 * 64 * 4 + 256 * 4 + 16 * 4;
    k_attention<<<dim3(SEQ / 16, BATCH * NUM_HEADS), 256, smem, stream>>>(q_bf, k_bf, v_bf, a_bf);

    // 4) output projection -> fp32
    k_gemm_wmma<<<512, 256, 0, stream>>>(a_bf, D_MODEL, Wo_bf, D_MODEL, out1,
                                         16, D_MODEL, D_MODEL, MODE_F32);

    // 5) x2 = LN(x + attn_out)
    k_resid_ln<<<MTOK, 256, 0, stream>>>(x, out1, g1, b1, x2f, x2_bf);

    // 6) FFN1 + exact GELU -> bf16 [4096][4096]; 32*64 = 2048 blocks
    k_gemm_wmma<<<2048, 256, 0, stream>>>(x2_bf, D_MODEL, W1_bf, D_MODEL, ff_bf,
                                          64, D_MODEL, D_FF, MODE_GELU_BF16);

    // 7) FFN2 -> fp32
    k_gemm_wmma<<<512, 256, 0, stream>>>(ff_bf, D_FF, W2_bf, D_FF, out2,
                                         16, D_FF, D_MODEL, MODE_F32);

    // 8) out = LN(x2 + ff_out)
    k_resid_ln<<<MTOK, 256, 0, stream>>>(x2f, out2, g2, b2, (float*)d_out, nullptr);
}
